// GCN_197568496255
// MI455X (gfx1250) — matrix operand order repaired and checked by reading, not compiled, not run on hardware
//
#include <hip/hip_runtime.h>
#include <stdint.h>

// ---------------------------------------------------------------------------
// GCN (2x GCNConv + mean-pool + linear) for MI455X / gfx1250, wave32.
// - Scatter phases: f32 + global_atomic_add_f32 (all buffers L2-resident).
// - N x 32 @ 32 x 32 GEMM: V_WMMA_F32_16X16X32_F16 with hi/lo f16 error
//   compensation (acc = Ah*Bh + Al*Bh + Ah*Bl, dropped term ~2^-22 rel).
// - W2 staged into LDS via the Tensor Data Mover (tensor_load_to_lds +
//   s_wait_tensorcnt), inline asm per CDNA5 ISA.
// ---------------------------------------------------------------------------

typedef __attribute__((ext_vector_type(16))) _Float16 v16h;
typedef __attribute__((ext_vector_type(8)))  float    v8f;
typedef __attribute__((ext_vector_type(4)))  unsigned v4u;
typedef __attribute__((ext_vector_type(8)))  unsigned v8u;

#define HID 32

__global__ void gcn_zero_f32(float* __restrict__ p, long long n) {
    long long i = (long long)blockIdx.x * blockDim.x + threadIdx.x;
    if (i < n) p[i] = 0.0f;
}

// deg[dst] += 1 for every edge (self-loop +1 folded into rsqrt kernel)
__global__ void gcn_degree(const int* __restrict__ dst, int E, float* __restrict__ deg) {
    int e = blockIdx.x * blockDim.x + threadIdx.x;
    if (e < E) atomicAdd(&deg[dst[e]], 1.0f);
}

// deg -> dinv = rsqrt(deg + 1)
__global__ void gcn_dinv(float* __restrict__ deg, int N) {
    int i = blockIdx.x * blockDim.x + threadIdx.x;
    if (i < N) deg[i] = rsqrtf(deg[i] + 1.0f);
}

// h = x @ W1   (N x 5 @ 5 x 32); one lane per (node, channel); x row is
// wave-uniform (scalar-cache friendly), W1/store fully coalesced.
__global__ void gcn_gemm1(const float* __restrict__ x, const float* __restrict__ W1,
                          float* __restrict__ h, int N) {
    long long t = (long long)blockIdx.x * blockDim.x + threadIdx.x;
    int n = (int)(t >> 5), j = (int)(t & 31);
    if (n >= N) return;
    const float* xr = x + (long long)n * 5;
    float s = 0.0f;
#pragma unroll
    for (int k = 0; k < 5; ++k) s += xr[k] * W1[k * HID + j];
    h[(long long)n * HID + j] = s;
}

// agg[dst] += h[src] * dinv[src]*dinv[dst]; lane j handles channel j of edge e.
// 128B coalesced gather per edge, f32 atomic scatter (L2 RMW).
__global__ void gcn_scatter(const int* __restrict__ src, const int* __restrict__ dst,
                            const float* __restrict__ dinv, const float* __restrict__ h,
                            float* __restrict__ agg, long long E) {
    long long t = (long long)blockIdx.x * blockDim.x + threadIdx.x;
    long long e = t >> 5;
    int j = (int)(t & 31);
    if (e >= E) return;
    int s = src[e], d = dst[e];
    float norm = dinv[s] * dinv[d];
    atomicAdd(&agg[(long long)d * HID + j], h[(long long)s * HID + j] * norm);
}

// out (in-place in agg) = [relu](agg + h * dinv^2 + b)
__global__ void gcn_combine(const float* __restrict__ h, const float* __restrict__ dinv,
                            const float* __restrict__ b, float* __restrict__ agg,
                            int N, int doRelu) {
    long long t = (long long)blockIdx.x * blockDim.x + threadIdx.x;
    int n = (int)(t >> 5), j = (int)(t & 31);
    if (n >= N) return;
    float di = dinv[n];
    float v = agg[(long long)n * HID + j] + h[(long long)n * HID + j] * di * di + b[j];
    if (doRelu) v = fmaxf(v, 0.0f);
    agg[(long long)n * HID + j] = v;
}

// h2 = out1 @ W2 (N x 32 @ 32 x 32) via v_wmma_f32_16x16x32_f16.
// Error-compensated: A,B split into f16 hi + lo; acc = Ah*Bh + Al*Bh + Ah*Bl.
// One wave per 16-node tile; W2 staged to LDS by the Tensor Data Mover.
__global__ void gcn_gemm2_wmma(const float* __restrict__ in, const float* __restrict__ W,
                               float* __restrict__ out, int N) {
    __shared__ float    sWf[HID * HID];     // raw f32 W2, filled by TDM
    __shared__ _Float16 sBhi[HID * HID];
    __shared__ _Float16 sBlo[HID * HID];
    int tid = threadIdx.x;

    // ---- TDM: DMA the 32x32 f32 tile W -> LDS (wave 0 only) ----
    if (tid < 32) {
        unsigned lds_off = (unsigned)(uintptr_t)(void*)sWf;     // addr[31:0] = LDS byte offset
        unsigned long long ga = (unsigned long long)(uintptr_t)W;
        // D# group0: [1:0]=count=1 | lds_addr | global_addr[56:0] | type=2
        v4u g0 = { 1u, lds_off, (unsigned)ga,
                   (unsigned)((ga >> 32) & 0x01FFFFFFu) | 0x80000000u };
        // D# group1: data_size=4B(code 2); tensor_dim0=32; tensor_dim1=32;
        //            tile_dim0=32; tile_dim1=32; tensor_dim0_stride=32
        v8u g1 = { 0x00020000u,      // workgroup_mask=0, data_size=2
                   32u << 16,        // bits63:48  = tensor_dim0.lo16 = 32
                   32u << 16,        // bits95:80  = tensor_dim1.lo16 = 32
                   32u << 16,        // bits127:112= tile_dim0 = 32
                   32u,              // bits143:128= tile_dim1 = 32
                   32u,              // bits191:160= tensor_dim0_stride.lo32 = 32
                   0u, 0u };
        asm volatile("tensor_load_to_lds %0, %1" :: "s"(g0), "s"(g1) : "memory");
        asm volatile("s_wait_tensorcnt 0x0" ::: "memory");
    }
    __syncthreads();

    // ---- split W into f16 hi/lo in LDS ----
    for (int i = tid; i < HID * HID; i += blockDim.x) {
        float w = sWf[i];
        _Float16 hi = (_Float16)w;
        sBhi[i] = hi;
        sBlo[i] = (_Float16)(w - (float)hi);
    }
    __syncthreads();

    int lane = tid & 31;
    int wave = (int)(((long long)blockIdx.x * blockDim.x + tid) >> 5); // tile id
    int numTiles = (N + 15) >> 4;
    if (wave >= numTiles) return;

    int row0 = wave << 4;
    int r    = row0 + (lane & 15);
    int rc   = (r < N) ? r : (N - 1);          // clamp tail; stores guarded below
    const float4* arow4 = (const float4*)(in + (long long)rc * HID);

    // A 16x32 f16 layout: lanes 0-15 hold K in {0..7,16..23}, lanes 16-31
    // hold K in {8..15,24..31}; element j -> K = kbA + (j<8 ? j : 8+j).
    int kq = (lane < 16) ? 0 : 2;              // kbA/4
    float4 fa = arow4[kq], fb = arow4[kq + 1], fc = arow4[kq + 4], fd = arow4[kq + 5];
    float ar[16] = { fa.x, fa.y, fa.z, fa.w, fb.x, fb.y, fb.z, fb.w,
                     fc.x, fc.y, fc.z, fc.w, fd.x, fd.y, fd.z, fd.w };
    v16h ahi, alo;
#pragma unroll
    for (int j = 0; j < 16; ++j) {
        float v = ar[j];
        _Float16 h = (_Float16)v;
        ahi[j] = h;
        alo[j] = (_Float16)(v - (float)h);
    }

    // B 32x16 f16 layout: lane holds column n = lane&15; lanes 0-15 K in {0..7,16..23},
    // lanes 16-31 K in {8..15,24..31}; element j -> K = kbB + (j<8 ? j : 8+j).
    int kbB   = (lane < 16) ? 0 : 8;
    int ncol  = lane & 15;
    int mbase = row0 + ((lane >> 4) << 3);     // D: VGPR v -> M = v (+8 hi half)

#pragma unroll
    for (int c0 = 0; c0 < 2; ++c0) {           // two 16-wide column tiles of W2
        v16h bhi, blo;
#pragma unroll
        for (int j = 0; j < 16; ++j) {
            int kk = kbB + (j < 8 ? j : 8 + j);
            bhi[j] = sBhi[kk * HID + c0 * 16 + ncol];
            blo[j] = sBlo[kk * HID + c0 * 16 + ncol];
        }
        v8f acc = {};
        acc = __builtin_amdgcn_wmma_f32_16x16x32_f16(false, ahi, false, bhi,
                                                     (short)0, acc, false, false);
        acc = __builtin_amdgcn_wmma_f32_16x16x32_f16(false, alo, false, bhi,
                                                     (short)0, acc, false, false);
        acc = __builtin_amdgcn_wmma_f32_16x16x32_f16(false, ahi, false, blo,
                                                     (short)0, acc, false, false);

        float* orow = out + (long long)mbase * HID + c0 * 16 + ncol;
        if (row0 + 16 <= N) {                  // uniform: full tile, branch-free stores
#pragma unroll
            for (int v = 0; v < 8; ++v) orow[v * HID] = acc[v];
        } else {                               // uniform: tail tile only
#pragma unroll
            for (int v = 0; v < 8; ++v)
                if (mbase + v < N) orow[v * HID] = acc[v];
        }
    }
}

// Final conv-2 combine fused with global mean-pool accumulation.
__global__ void gcn_combine_pool(const float* __restrict__ h, const float* __restrict__ dinv,
                                 const float* __restrict__ b, const float* __restrict__ agg,
                                 const int* __restrict__ batch, float* __restrict__ pooled,
                                 float* __restrict__ cnt, int N) {
    long long t = (long long)blockIdx.x * blockDim.x + threadIdx.x;
    int n = (int)(t >> 5), j = (int)(t & 31);
    if (n >= N) return;
    float di = dinv[n];
    float v = agg[(long long)n * HID + j] + h[(long long)n * HID + j] * di * di + b[j];
    int g = batch[n];
    atomicAdd(&pooled[(long long)g * HID + j], v);
    if (j == 0) atomicAdd(&cnt[g], 1.0f);
}

// out[g,c] = (pooled[g,:]/max(cnt,1)) @ Wfc[:,c] + bfc[c]
__global__ void gcn_head(const float* __restrict__ pooled, const float* __restrict__ cnt,
                         const float* __restrict__ Wfc, const float* __restrict__ bfc,
                         float* __restrict__ out, int G, int C) {
    int t = blockIdx.x * blockDim.x + threadIdx.x;
    if (t >= G * C) return;
    int g = t / C, c = t % C;
    float inv = 1.0f / fmaxf(cnt[g], 1.0f);
    float s = 0.0f;
#pragma unroll
    for (int k = 0; k < HID; ++k) s += pooled[g * HID + k] * Wfc[k * C + c];
    out[t] = s * inv + bfc[c];
}

static inline unsigned cdiv(long long a, long long b) { return (unsigned)((a + b - 1) / b); }

extern "C" void kernel_launch(void* const* d_in, const int* in_sizes, int n_in,
                              void* d_out, int out_size, void* d_ws, size_t ws_size,
                              hipStream_t stream) {
    const float* x   = (const float*)d_in[0];
    const float* W1  = (const float*)d_in[1];
    const float* b1  = (const float*)d_in[2];
    const float* W2  = (const float*)d_in[3];
    const float* b2  = (const float*)d_in[4];
    const float* Wfc = (const float*)d_in[5];
    const float* bfc = (const float*)d_in[6];
    const int*   ei  = (const int*)d_in[7];   // [2, E] flat
    const int*   bat = (const int*)d_in[8];   // [N]
    float* out = (float*)d_out;

    const int N = in_sizes[0] / 5;
    const long long E = in_sizes[7] / 2;
    const int C = 2;
    const int G = out_size / C;
    const int* src = ei;
    const int* dst = ei + E;

    // workspace carve-up (floats)
    float* ws     = (float*)d_ws;
    float* dinv   = ws;                               // N   (also deg accumulator)
    float* bufA   = dinv + N;                         // N*32 (h per layer)
    float* bufB   = bufA + (long long)N * HID;        // N*32 (agg / out per layer)
    float* pooled = bufB + (long long)N * HID;        // G*32
    float* cnt    = pooled + (long long)G * HID;      // G
    (void)ws_size; (void)n_in;

    const int B = 256;
    const long long NH = (long long)N * HID;

    // degrees -> dinv ; zero pool accumulators
    gcn_zero_f32<<<cdiv(N, B), B, 0, stream>>>(dinv, N);
    gcn_zero_f32<<<cdiv((long long)G * (HID + 1), B), B, 0, stream>>>(pooled, (long long)G * (HID + 1));
    gcn_degree<<<cdiv(E, B), B, 0, stream>>>(dst, (int)E, dinv);
    gcn_dinv<<<cdiv(N, B), B, 0, stream>>>(dinv, N);

    // layer 1: h1 = x@W1 ; agg ; relu(agg + h*dinv^2 + b1) -> bufB
    gcn_gemm1<<<cdiv((long long)N * HID, B), B, 0, stream>>>(x, W1, bufA, N);
    gcn_zero_f32<<<cdiv(NH, B), B, 0, stream>>>(bufB, NH);
    gcn_scatter<<<cdiv(E * HID, B), B, 0, stream>>>(src, dst, dinv, bufA, bufB, E);
    gcn_combine<<<cdiv(NH, B), B, 0, stream>>>(bufA, dinv, b1, bufB, N, 1);

    // layer 2: h2 = out1@W2 (WMMA + TDM) ; agg ; combine fused with mean-pool
    {
        int numTiles = (N + 15) >> 4;                 // one wave per 16-row tile
        unsigned blocks = cdiv((long long)numTiles * 32, B);
        gcn_gemm2_wmma<<<blocks, B, 0, stream>>>(bufB, W2, bufA, N);
    }
    gcn_zero_f32<<<cdiv(NH, B), B, 0, stream>>>(bufB, NH);
    gcn_scatter<<<cdiv(E * HID, B), B, 0, stream>>>(src, dst, dinv, bufA, bufB, E);
    gcn_combine_pool<<<cdiv(NH, B), B, 0, stream>>>(bufA, dinv, b2, bufB, bat, pooled, cnt, N);

    // head
    gcn_head<<<cdiv((long long)G * C, B), B, 0, stream>>>(pooled, cnt, Wfc, bfc, out, G, C);
}
